// TNformer_MP_46703474376984
// MI455X (gfx1250) — compile-verified, matrix operand order checked by reading, not run on hardware
//
#include <hip/hip_runtime.h>
#include <math.h>

// Problem constants (match reference)
#define B_      16
#define M_      128
#define L_      512
#define HID_    256
#define H_      4
#define DH_     64
#define LAYERS_ 2

typedef __attribute__((ext_vector_type(16))) _Float16 v16h;
typedef __attribute__((ext_vector_type(8)))  _Float16 v8h;
typedef __attribute__((ext_vector_type(2)))  _Float16 v2h;
typedef __attribute__((ext_vector_type(8)))  float    v8f;

// ---------------------------------------------------------------------------
// Generic batched WMMA GEMM:  C = act(alpha * A @ op(B) + bias + resid)
//   A     : f32 [M,K], leading dim lda
//   B     : f32, transB ? [N,K] (ldb = row stride over N) : [K,N]
//   C/res : f32 [M,N], leading dim ldc (resid uses C's layout & batch strides)
//   batch : grid.z = nB*nH ; z -> (b = z/nH, h = z%nH), element strides per dim
// Requires M%128==0, N%64==0, K%32==0 (true for every call below).
// Block: 128 threads = 4 wave32. Block tile 128x64, each wave 32x64
// (2 A-frags x 4 B-frags = 8 v_wmma per K-step of 32).
// LDS tiles are stored so each lane's 16-half fragment is two aligned 16-byte
// runs -> fragments load as two ds_load_b128 (A layout: lanes0-15 K=hi*8+0..7
// then K=16+hi*8+0..7; B tile stored N-major so it gathers identically).
// ---------------------------------------------------------------------------
__global__ __launch_bounds__(128)
void gemm_f16_wmma(const float* __restrict__ A, const float* __restrict__ Bm,
                   const float* __restrict__ bias, const float* __restrict__ resid,
                   float* __restrict__ C,
                   int M, int N, int K, int lda, int ldb, int ldc,
                   long sAb, long sAh, long sBb, long sBh, long sCb, long sCh,
                   int nH, int transB, float alpha, int act)
{
    __shared__ __attribute__((aligned(16))) _Float16 lsA[128][40]; // [m][k], 80B rows
    __shared__ __attribute__((aligned(16))) _Float16 lsB[64][40];  // [n][k], 80B rows

    const int z  = blockIdx.z;
    const int bb = z / nH, hh = z % nH;
    A  += (long)bb * sAb + (long)hh * sAh;
    Bm += (long)bb * sBb + (long)hh * sBh;
    const long coff = (long)bb * sCb + (long)hh * sCh;
    C += coff;
    if (resid) resid += coff;

    const int tid  = threadIdx.x;
    const int wave = tid >> 5;                 // 4 waves stacked along M
    const int lane = tid & 31;
    const int row0 = blockIdx.y * 128;
    const int col0 = blockIdx.x * 64;
    const int hi = lane >> 4, lm = lane & 15;

    v8f acc[2][4] = {};

    for (int k0 = 0; k0 < K; k0 += 32) {
        // ---- Stage A tile 128x32 (f32 -> f16), paired converts, b32 stores
        for (int e = tid; e < 128 * 16; e += 128) {          // float2 units
            int r = e >> 4, k2 = e & 15;
            float2 f = *(const float2*)&A[(long)(row0 + r) * lda + (k0 + 2 * k2)];
            v2h p; p[0] = (_Float16)f.x; p[1] = (_Float16)f.y;
            *(v2h*)&lsA[r][2 * k2] = p;
        }
        // ---- Stage B tile into [n][k]
        if (transB) {
            for (int e = tid; e < 64 * 16; e += 128) {       // float2 units
                int n = e >> 4, k2 = e & 15;
                float2 f = *(const float2*)&Bm[(long)(col0 + n) * ldb + (k0 + 2 * k2)];
                v2h p; p[0] = (_Float16)f.x; p[1] = (_Float16)f.y;
                *(v2h*)&lsB[n][2 * k2] = p;
            }
        } else {
            for (int e = tid; e < 64 * 32; e += 128) {       // coalesced read over n
                int k = e >> 6, n = e & 63;
                lsB[n][k] = (_Float16)Bm[(long)(k0 + k) * ldb + (col0 + n)];
            }
        }
        // Prefetch next K-tile (global_prefetch_b8) to overlap with WMMA phase
        if (k0 + 32 < K) {
            __builtin_prefetch(&A[(long)(row0 + tid) * lda + (k0 + 32)], 0, 1);
            if (tid < 64)
                __builtin_prefetch(transB
                        ? &Bm[(long)(col0 + tid) * ldb + (k0 + 32)]
                        : &Bm[(long)(k0 + 32 + (tid >> 5)) * ldb + (col0 + (tid & 31) * 2)], 0, 1);
        }
        __syncthreads();

        // ---- Fragments: two aligned 16B runs per lane -> ds_load_b128 pairs
        v16h af[2], bf[4];
#pragma unroll
        for (int fa = 0; fa < 2; ++fa) {
            const int ra = wave * 32 + fa * 16 + lm;
            v8h lo = *(const v8h*)&lsA[ra][hi * 8];
            v8h hi8 = *(const v8h*)&lsA[ra][16 + hi * 8];
            af[fa] = __builtin_shufflevector(lo, hi8, 0,1,2,3,4,5,6,7,8,9,10,11,12,13,14,15);
        }
#pragma unroll
        for (int fb = 0; fb < 4; ++fb) {
            const int cb = fb * 16 + lm;
            v8h lo = *(const v8h*)&lsB[cb][hi * 8];
            v8h hi8 = *(const v8h*)&lsB[cb][16 + hi * 8];
            bf[fb] = __builtin_shufflevector(lo, hi8, 0,1,2,3,4,5,6,7,8,9,10,11,12,13,14,15);
        }
#pragma unroll
        for (int fa = 0; fa < 2; ++fa)
#pragma unroll
            for (int fb = 0; fb < 4; ++fb)
                acc[fa][fb] = __builtin_amdgcn_wmma_f32_16x16x32_f16(
                    false, af[fa], false, bf[fb], (short)0, acc[fa][fb], false, false);
        __syncthreads();
    }

    // Epilogue. C/D layout: VGPR i -> row = i + 8*hi (within 16x16), col = lane&15.
#pragma unroll
    for (int fa = 0; fa < 2; ++fa) {
#pragma unroll
        for (int fb = 0; fb < 4; ++fb) {
#pragma unroll
            for (int i = 0; i < 8; ++i) {
                int r = row0 + wave * 32 + fa * 16 + hi * 8 + i;
                int c = col0 + fb * 16 + lm;
                float v = acc[fa][fb][i] * alpha;
                if (bias)  v += bias[c];
                if (resid) v += resid[(long)r * ldc + c];
                if (act == 1)      v = (v > 0.f) ? v : (__expf(v) - 1.f);           // ELU
                else if (act == 2) v = 0.5f * v * (1.f + erff(v * 0.70710678118f)); // exact GELU
                C[(long)r * ldc + c] = v;
            }
        }
    }
}

// ---------------------------------------------------------------------------
// TN tokenizer: temp[b,n,s*64+d] = sum_m softmax_d(-s*alpha*t)[d] * ct_e[b,m,s*64+d]
// with t = (ct_t[b,m]-ts_t[b,n])^2, alpha = softplus(kernel_param).
// z = -s*alpha*t in [-6,0] so direct exp / sum-exp == reference logsumexp form.
// ---------------------------------------------------------------------------
__global__ __launch_bounds__(256)
void tokenizer_kernel(const float* __restrict__ ct_t, const float* __restrict__ ts_t,
                      const float* __restrict__ ct_e, const float* __restrict__ kparam,
                      float* __restrict__ temp)
{
    __shared__ float alpha_s[DH_];
    __shared__ float part[8];
    const int tid = threadIdx.x;
    const int n = blockIdx.x, b = blockIdx.y;
    const int s = tid >> 6, d = tid & 63;
    if (tid < DH_) alpha_s[tid] = log1pf(__expf(kparam[tid]));  // softplus
    __syncthreads();
    const float sa  = (float)s * alpha_s[d];
    const float tsn = ts_t[b * L_ + n];
    float acc = 0.f;
    for (int m = 0; m < M_; ++m) {
        float df = ct_t[b * M_ + m] - tsn;
        float e  = __expf(-sa * df * df);
        float r  = e;
#pragma unroll
        for (int off = 16; off > 0; off >>= 1) r += __shfl_xor(r, off, 32);
        if ((tid & 31) == 0) part[tid >> 5] = r;
        __syncthreads();
        float Z = part[2 * s] + part[2 * s + 1];
        acc += (e / Z) * ct_e[((long)(b * M_ + m)) * HID_ + tid];
        __syncthreads();
    }
    temp[((long)(b * L_ + n)) * HID_ + tid] = acc;
}

// ---------------------------------------------------------------------------
// Softmax over rows of 512 (scores pre-scaled by 1/sqrt(DH) via GEMM alpha)
// ---------------------------------------------------------------------------
__global__ __launch_bounds__(256)
void softmax512_kernel(float* __restrict__ S)
{
    __shared__ float sm[256];
    const int tid = threadIdx.x;
    float* row = S + (long)blockIdx.x * 512;
    float a = row[tid], b = row[tid + 256];
    sm[tid] = fmaxf(a, b); __syncthreads();
    for (int s = 128; s > 0; s >>= 1) { if (tid < s) sm[tid] = fmaxf(sm[tid], sm[tid + s]); __syncthreads(); }
    float mx = sm[0]; __syncthreads();
    float ea = __expf(a - mx), eb = __expf(b - mx);
    sm[tid] = ea + eb; __syncthreads();
    for (int s = 128; s > 0; s >>= 1) { if (tid < s) sm[tid] += sm[tid + s]; __syncthreads(); }
    float inv = 1.f / sm[0];
    row[tid] = ea * inv; row[tid + 256] = eb * inv;
}

// ---------------------------------------------------------------------------
// LayerNorm over rows of HID_ (=256), eps = 1e-5
// ---------------------------------------------------------------------------
__global__ __launch_bounds__(256)
void layernorm_kernel(const float* __restrict__ X, const float* __restrict__ g,
                      const float* __restrict__ bta, float* __restrict__ Y)
{
    __shared__ float sm[256];
    const int tid = threadIdx.x;
    const long base = (long)blockIdx.x * HID_;
    float v = X[base + tid];
    sm[tid] = v; __syncthreads();
    for (int s = 128; s > 0; s >>= 1) { if (tid < s) sm[tid] += sm[tid + s]; __syncthreads(); }
    float mu = sm[0] * (1.f / HID_); __syncthreads();
    float t = v - mu;
    sm[tid] = t * t; __syncthreads();
    for (int s = 128; s > 0; s >>= 1) { if (tid < s) sm[tid] += sm[tid + s]; __syncthreads(); }
    float var = sm[0] * (1.f / HID_);
    Y[base + tid] = t * rsqrtf(var + 1e-5f) * g[tid] + bta[tid];
}

// ---------------------------------------------------------------------------
// Prediction head: out[row] = dot(x[row,:], pred_W) + pred_b   (N=1 -> no WMMA)
// ---------------------------------------------------------------------------
__global__ __launch_bounds__(256)
void pred_kernel(const float* __restrict__ X, const float* __restrict__ W,
                 const float* __restrict__ bsc, float* __restrict__ out)
{
    __shared__ float sm[256];
    const int tid = threadIdx.x;
    const long base = (long)blockIdx.x * HID_;
    sm[tid] = X[base + tid] * W[tid]; __syncthreads();
    for (int s = 128; s > 0; s >>= 1) { if (tid < s) sm[tid] += sm[tid + s]; __syncthreads(); }
    if (tid == 0) out[blockIdx.x] = sm[0] + bsc[0];
}

// ---------------------------------------------------------------------------
static void launch_gemm(hipStream_t st, const float* A, const float* Bm,
                        const float* bias, const float* resid, float* C,
                        int M, int N, int K, int lda, int ldb, int ldc,
                        long sAb, long sAh, long sBb, long sBh, long sCb, long sCh,
                        int nB, int nH, int transB, float alpha, int act)
{
    dim3 grid(N / 64, M / 128, nB * nH);
    gemm_f16_wmma<<<grid, dim3(128), 0, st>>>(A, Bm, bias, resid, C,
                                              M, N, K, lda, ldb, ldc,
                                              sAb, sAh, sBb, sBh, sCb, sCh,
                                              nH, transB, alpha, act);
}

extern "C" void kernel_launch(void* const* d_in, const int* in_sizes, int n_in,
                              void* d_out, int out_size, void* d_ws, size_t ws_size,
                              hipStream_t stream)
{
    (void)in_sizes; (void)n_in; (void)out_size; (void)ws_size;
    const float* ct_t   = (const float*)d_in[0];
    const float* ts_t   = (const float*)d_in[1];
    const float* ct_emb = (const float*)d_in[2];
    const float* ts_emb = (const float*)d_in[3];
    const float* kparam = (const float*)d_in[4];
    const float* ct_W   = (const float*)d_in[5];
    const float* ct_b   = (const float*)d_in[6];
    const float* fW0    = (const float*)d_in[7];
    const float* fb0    = (const float*)d_in[8];
    const float* fW1    = (const float*)d_in[9];
    const float* fb1    = (const float*)d_in[10];
    const float* fW2    = (const float*)d_in[11];
    const float* fb2    = (const float*)d_in[12];
    const float* f_ln_g = (const float*)d_in[13];
    const float* f_ln_b = (const float*)d_in[14];
    const float* Wqkv   = (const float*)d_in[15];
    const float* bqkv   = (const float*)d_in[16];
    const float* Wo     = (const float*)d_in[17];
    const float* bo     = (const float*)d_in[18];
    const float* W1w    = (const float*)d_in[19];
    const float* b1v    = (const float*)d_in[20];
    const float* W2w    = (const float*)d_in[21];
    const float* b2v    = (const float*)d_in[22];
    const float* ln1_g  = (const float*)d_in[23];
    const float* ln1_b  = (const float*)d_in[24];
    const float* ln2_g  = (const float*)d_in[25];
    const float* ln2_b  = (const float*)d_in[26];
    const float* pred_W = (const float*)d_in[27];
    const float* pred_b = (const float*)d_in[28];
    float* out = (float*)d_out;

    // Workspace layout (floats). Peak ~136 MB -- lives in the 192 MB L2.
    float* wsf    = (float*)d_ws;
    float* ct_e   = wsf;              //  2048*256          =   524288
    float* temp   = wsf +   524288;   //  8192*256          =  2097152
    float* bufA   = wsf +  2621440;   //  8192*256
    float* aux    = wsf +  4718592;   //  8192*256  (reused as attention O buffer)
    float* bufB   = wsf +  6815744;   //  8192*256  (pre-LN scratch)
    float* x      = wsf +  8912896;   //  8192*256
    float* qkv    = wsf + 11010048;   //  8192*768          =  6291456
    float* scores = wsf + 17301504;   //  64*512*512        = 16777216 (reused as FF hidden)
    float* obuf   = aux;

    const int BM = B_ * M_;   // 2048
    const int BL = B_ * L_;   // 8192

    // --- TN tokenizer ---
    // ct_e = ct_emb @ ct_W^T + ct_b   ('bmd,ed->bme')
    launch_gemm(stream, ct_emb, ct_W, ct_b, nullptr, ct_e,
                BM, HID_, HID_, HID_, HID_, HID_, 0,0,0,0,0,0, 1,1, 1, 1.f, 0);
    tokenizer_kernel<<<dim3(L_, B_), 256, 0, stream>>>(ct_t, ts_t, ct_e, kparam, temp);

    // --- Fusion ---
    launch_gemm(stream, temp,   fW0, fb0, nullptr, bufA,
                BL, HID_, HID_, HID_, HID_, HID_, 0,0,0,0,0,0, 1,1, 1, 1.f, 0);
    launch_gemm(stream, ts_emb, fW1, fb1, bufA,    aux,        // ELU(temp@fW0+ts@fW1+b)
                BL, HID_, HID_, HID_, HID_, HID_, 0,0,0,0,0,0, 1,1, 1, 1.f, 1);
    launch_gemm(stream, aux,    fW2, fb2, ts_emb,  bufB,       // + ts_emb residual
                BL, HID_, HID_, HID_, HID_, HID_, 0,0,0,0,0,0, 1,1, 1, 1.f, 0);
    layernorm_kernel<<<BL, 256, 0, stream>>>(bufB, f_ln_g, f_ln_b, x);

    // --- Transformer encoder ---
    const float inv_sqrt = 0.125f;  // 1/sqrt(64)
    for (int l = 0; l < LAYERS_; ++l) {
        // qkv = x @ Wqkv[l]^T + bqkv[l]
        launch_gemm(stream, x, Wqkv + (long)l * 3 * HID_ * HID_,
                    bqkv + (long)l * 3 * HID_, nullptr, qkv,
                    BL, 3 * HID_, HID_, HID_, HID_, 3 * HID_, 0,0,0,0,0,0, 1,1, 1, 1.f, 0);

        // scores[b,h] = (Q @ K^T) / sqrt(DH)    batched over (b,h)
        launch_gemm(stream, qkv /*Q*/, qkv + HID_ /*K*/, nullptr, nullptr, scores,
                    L_, L_, DH_, 3 * HID_, 3 * HID_, L_,
                    (long)L_ * 3 * HID_, DH_,            // A strides (b,h)
                    (long)L_ * 3 * HID_, DH_,            // B strides (b,h)
                    (long)H_ * L_ * L_, (long)L_ * L_,   // C strides (b,h)
                    B_, H_, 1, inv_sqrt, 0);
        softmax512_kernel<<<B_ * H_ * L_, 256, 0, stream>>>(scores);

        // O[b,:,h*64:...] = P @ V
        launch_gemm(stream, scores, qkv + 2 * HID_ /*V*/, nullptr, nullptr, obuf,
                    L_, DH_, L_, L_, 3 * HID_, HID_,
                    (long)H_ * L_ * L_, (long)L_ * L_,   // A strides (b,h)
                    (long)L_ * 3 * HID_, DH_,            // B strides (b,h)
                    (long)L_ * HID_, DH_,                // C strides (b,h)
                    B_, H_, 0, 1.f, 0);

        // x = LN(x + O @ Wo^T + bo)
        launch_gemm(stream, obuf, Wo + (long)l * HID_ * HID_,
                    bo + (long)l * HID_, x, bufB,
                    BL, HID_, HID_, HID_, HID_, HID_, 0,0,0,0,0,0, 1,1, 1, 1.f, 0);
        layernorm_kernel<<<BL, 256, 0, stream>>>(bufB, ln1_g + l * HID_, ln1_b + l * HID_, x);

        // FF: x = LN(x + GELU(x@W1^T+b1) @ W2^T + b2)
        float* ffh = scores;  // reuse scores region (8192*512 floats)
        launch_gemm(stream, x, W1w + (long)l * 2 * HID_ * HID_,
                    b1v + (long)l * 2 * HID_, nullptr, ffh,
                    BL, 2 * HID_, HID_, HID_, HID_, 2 * HID_, 0,0,0,0,0,0, 1,1, 1, 1.f, 2);
        launch_gemm(stream, ffh, W2w + (long)l * HID_ * 2 * HID_,
                    b2v + (long)l * HID_, x, bufB,
                    BL, HID_, 2 * HID_, 2 * HID_, 2 * HID_, HID_, 0,0,0,0,0,0, 1,1, 1, 1.f, 0);
        layernorm_kernel<<<BL, 256, 0, stream>>>(bufB, ln2_g + l * HID_, ln2_b + l * HID_, x);
    }

    // --- Prediction head ---
    pred_kernel<<<BL, 256, 0, stream>>>(x, pred_W, pred_b, out);
}
